// FeatureCorrelator_2147483648362
// MI455X (gfx1250) — matrix operand order repaired and checked by reading, not compiled, not run on hardware
//
#include <hip/hip_runtime.h>
#include <cstdint>
#include <cstddef>

typedef __attribute__((ext_vector_type(16))) _Float16 v16h;
typedef __attribute__((ext_vector_type(8)))  float    v8f;

#define BB   2
#define NN   4096
#define DD   64
#define KNS  16
#define KCL  8
#define FDIM 67
#define FPAD 68
#define CIN  131
#define CINP 160
#define HD   128

__device__ __forceinline__ float reluf(float x)  { return x > 0.f ? x : 0.f; }
__device__ __forceinline__ float leakyf(float x) { return x > 0.f ? x : 0.1f * x; }

// ---------------------------------------------------------------------------
// 1) Transpose + scale inputs: build ft1/ft2 [B*N,68], x1t/x2t [B*N,4]
// ---------------------------------------------------------------------------
__global__ void prep_kernel(const float* __restrict__ xyz1, const float* __restrict__ xyz2,
                            const float* __restrict__ p1,   const float* __restrict__ p2,
                            const float* __restrict__ wxyz, const float* __restrict__ wpts,
                            float* __restrict__ ft1, float* __restrict__ ft2,
                            float* __restrict__ x1t, float* __restrict__ x2t) {
  int g = blockIdx.x * blockDim.x + threadIdx.x;
  if (g >= BB * NN) return;
  int b = g / NN, n = g % NN;
  float wx = wxyz[0], wp = wpts[0];
  const float* X1 = xyz1 + (size_t)b * 3 * NN;
  const float* X2 = xyz2 + (size_t)b * 3 * NN;
  const float* P1 = p1 + (size_t)b * DD * NN;
  const float* P2 = p2 + (size_t)b * DD * NN;
#pragma unroll
  for (int c = 0; c < 3; ++c) {
    float a = X1[c * NN + n], d = X2[c * NN + n];
    x1t[g * 4 + c] = a;       x2t[g * 4 + c] = d;
    ft1[g * FPAD + c] = wx * a; ft2[g * FPAD + c] = wx * d;
  }
  x1t[g * 4 + 3] = 0.f; x2t[g * 4 + 3] = 0.f;
  for (int c = 0; c < DD; ++c) {
    ft1[g * FPAD + 3 + c] = wp * P1[c * NN + n];
    ft2[g * FPAD + 3 + c] = wp * P2[c * NN + n];
  }
  ft1[g * FPAD + FDIM] = 0.f; ft2[g * FPAD + FDIM] = 0.f;
}

// ---------------------------------------------------------------------------
// 2) Convert MLP weights to f16, pre-swizzled into WMMA B-fragment order.
//    B frag (16x16x32 f16): lane 0-15 -> N=lane, K = kk*32 + e (e=0..15)
//                           lane16-31 -> N=lane-16, K = kk*32 + 16 + e
//    w1s: 8 N-tiles x 5 K-chunks, w2s: 8 x 4. 512 halfs per fragment.
// ---------------------------------------------------------------------------
#define W1TOT (8 * 5 * 512)
#define W2TOT (8 * 4 * 512)
__global__ void swizzle_w_kernel(const float* __restrict__ w1, const float* __restrict__ w2,
                                 _Float16* __restrict__ w1s, _Float16* __restrict__ w2s) {
  int t = blockIdx.x * blockDim.x + threadIdx.x;
  if (t < W1TOT) {
    int e = t & 15, lane = (t >> 4) & 31, frag = t >> 9;
    int kk = frag % 5, nt = frag / 5;
    int K = kk * 32 + ((lane >= 16) ? 16 : 0) + e;
    int col = nt * 16 + (lane & 15);
    w1s[t] = (K < CIN) ? (_Float16)w1[K * HD + col] : (_Float16)0.f;
  } else if (t < W1TOT + W2TOT) {
    int t2 = t - W1TOT;
    int e = t2 & 15, lane = (t2 >> 4) & 31, frag = t2 >> 9;
    int kk = frag % 4, nt = frag / 4;
    int K = kk * 32 + ((lane >= 16) ? 16 : 0) + e;
    int col = nt * 16 + (lane & 15);
    w2s[t2] = (_Float16)w2[K * HD + col];
  }
}

// ---------------------------------------------------------------------------
// 3) Fused distance + top-k (tie-break low index). One block per query row.
// ---------------------------------------------------------------------------
__global__ void knn_kernel(const float* __restrict__ query, const float* __restrict__ data,
                           int fstride, int fdim, int k, int* __restrict__ out) {
  __shared__ float dist[NN];
  __shared__ float q[FPAD];
  __shared__ float sdist[256];
  __shared__ int   sidx[256];
  int t = threadIdx.x;
  int gi = blockIdx.x;              // b*NN + i
  int b = gi / NN;
  const float* qrow = query + (size_t)gi * fstride;
  const float* drow = data + (size_t)b * NN * fstride;
  if (t < fdim) q[t] = qrow[t];
  __syncthreads();
  for (int j = t; j < NN; j += 256) {
    const float* dj = drow + (size_t)j * fstride;
    __builtin_prefetch(dj + 256 * fstride, 0, 1);
    float s = 0.f;
    for (int c = 0; c < fdim; ++c) { float d = q[c] - dj[c]; s += d * d; }
    dist[j] = s;
  }
  __syncthreads();
  for (int it = 0; it < k; ++it) {
    float best = 3.4e38f; int bidx = NN;
    for (int j = t; j < NN; j += 256) {
      float d = dist[j];
      if (d < best) { best = d; bidx = j; }
    }
    sdist[t] = best; sidx[t] = bidx;
    __syncthreads();
    for (int s = 128; s > 0; s >>= 1) {
      if (t < s) {
        float d2 = sdist[t + s]; int i2 = sidx[t + s];
        if (d2 < sdist[t] || (d2 == sdist[t] && i2 < sidx[t])) { sdist[t] = d2; sidx[t] = i2; }
      }
      __syncthreads();
    }
    if (t == 0) { out[(size_t)gi * k + it] = sidx[0]; dist[sidx[0]] = 3.4e38f; }
    __syncthreads();
  }
}

// ---------------------------------------------------------------------------
// 4) Rigid velocity regression: 3x3 normal equations via Cramer, per point.
// ---------------------------------------------------------------------------
__global__ void vworld_kernel(const float* __restrict__ x1t, const float* __restrict__ vel1,
                              const int* __restrict__ idxc,
                              float* __restrict__ vw, float* __restrict__ out_tail) {
  int g = blockIdx.x * blockDim.x + threadIdx.x;
  if (g >= BB * NN) return;
  int b = g / NN;
  const float* X = x1t + (size_t)b * NN * 4;
  const float* V = vel1 + (size_t)b * NN;
  float a00 = 1e-6f, a01 = 0.f, a02 = 0.f, a11 = 1e-6f, a12 = 0.f, a22 = 1e-6f;
  float b0 = 0.f, b1 = 0.f, b2 = 0.f;
#pragma unroll
  for (int k = 0; k < KCL; ++k) {
    int id = idxc[(size_t)g * KCL + k];
    float x = X[id * 4 + 0], y = X[id * 4 + 1], z = X[id * 4 + 2];
    float inv = 1.f / sqrtf(x * x + y * y + z * z);
    float ux = x * inv, uy = y * inv, uz = z * inv;
    a00 += ux * ux; a01 += ux * uy; a02 += ux * uz;
    a11 += uy * uy; a12 += uy * uz; a22 += uz * uz;
    float cv = V[id];
    b0 += ux * cv; b1 += uy * cv; b2 += uz * cv;
  }
  float c00 = a11 * a22 - a12 * a12;
  float det = a00 * c00 - a01 * (a01 * a22 - a12 * a02) + a02 * (a01 * a12 - a11 * a02);
  float idet = 1.f / det;
  float v0 = (b0 * c00 - a01 * (b1 * a22 - a12 * b2) + a02 * (b1 * a12 - a11 * b2)) * idet;
  float v1 = (a00 * (b1 * a22 - a12 * b2) - b0 * (a01 * a22 - a12 * a02) + a02 * (a01 * b2 - b1 * a02)) * idet;
  float v2 = (a00 * (a11 * b2 - b1 * a12) - a01 * (a01 * b2 - b1 * a02) + b0 * (a01 * a12 - a11 * a02)) * idet;
  vw[g * 4 + 0] = v0; vw[g * 4 + 1] = v1; vw[g * 4 + 2] = v2; vw[g * 4 + 3] = 0.f;
  out_tail[(size_t)g * 3 + 0] = v0;
  out_tail[(size_t)g * 3 + 1] = v1;
  out_tail[(size_t)g * 3 + 2] = v2;
}

// ---------------------------------------------------------------------------
// 5) WMMA cost-volume MLP + WeightNet1 + K-reduction. One wave per point.
//    GEMM1: A[16x160(f16)] x W1s -> leaky -> LDS -> GEMM2: x W2s -> leaky.
//    point_to_patch[n,col] = sum_m relu(wn3(h2_m))[col] * H2[m,col]
// ---------------------------------------------------------------------------
__global__ __launch_bounds__(256) void mlp_ptp_kernel(
    const float* __restrict__ p1, const float* __restrict__ p2,
    const float* __restrict__ x1t, const float* __restrict__ x2t,
    const int* __restrict__ knn,
    const _Float16* __restrict__ w1s, const _Float16* __restrict__ w2s,
    const float* __restrict__ b1g, const float* __restrict__ b2g,
    const float* __restrict__ wn_w1, const float* __restrict__ wn_b1,
    const float* __restrict__ wn_w2, const float* __restrict__ wn_b2,
    const float* __restrict__ wn_w3, const float* __restrict__ wn_b3,
    float* __restrict__ ptp) {
  __shared__ _Float16 s_feat[8][16 * CINP];   // 40 KB
  __shared__ int      s_idx[8][16];
  __shared__ float    s_wnh[8][16][8];

  int tid = threadIdx.x;
  int w = tid >> 5;
  int lane = tid & 31;
  int g = blockIdx.x * 8 + w;                 // global point, grid = B*N/8
  int b = g / NN, n = g % NN;
  const float* P1 = p1 + (size_t)b * DD * NN;
  const float* P2 = p2 + (size_t)b * DD * NN;
  const float* X1 = x1t + (size_t)b * NN * 4;
  const float* X2 = x2t + (size_t)b * NN * 4;

  if (lane < 16) s_idx[w][lane] = knn[(size_t)g * KNS + lane];
  __syncthreads();

  // stage new_points rows [16 x 160] (f16): [p1(n) | p2(idx_m) | dir_m | 0pad]
  for (int t = lane; t < 16 * CINP; t += 32) {
    int m = t / CINP, c = t - m * CINP;
    float val;
    if (c < DD)            val = P1[c * NN + n];
    else if (c < 2 * DD)   val = P2[(c - DD) * NN + s_idx[w][m]];
    else if (c < CIN)      { int j = c - 2 * DD; val = X2[s_idx[w][m] * 4 + j] - X1[n * 4 + j]; }
    else                   val = 0.f;
    s_feat[w][t] = (_Float16)val;
  }
  __syncthreads();

  const int hi = (lane >= 16) ? 1 : 0;
  const int col0 = lane & 15;
  const int Mrow = lane & 15;

  // ---- GEMM1: 8 N-tiles x 5 K-chunks ----
  v8f acc[8];
#pragma unroll
  for (int nt = 0; nt < 8; ++nt) {
    float bv = b1g[nt * 16 + col0];
#pragma unroll
    for (int j = 0; j < 8; ++j) acc[nt][j] = bv;
  }
  const _Float16* fb = &s_feat[w][0];
#pragma unroll
  for (int kk = 0; kk < 5; ++kk) {
    v16h a;
#pragma unroll
    for (int e = 0; e < 16; ++e) {
      int vv = e >> 1, h = e & 1;
      int kl = (2 * vv + h) + 16 * hi;
      a[e] = fb[Mrow * CINP + kk * 32 + kl];
    }
#pragma unroll
    for (int nt = 0; nt < 8; ++nt) {
      v16h bw = *((const v16h*)w1s + (nt * 5 + kk) * 32 + lane);
      acc[nt] = __builtin_amdgcn_wmma_f32_16x16x32_f16(
          false, a, false, bw, (short)0, acc[nt], false, false);
    }
  }

  // leaky + restripe H1 into LDS as f16 [16 x 128] (A-layout source for GEMM2)
  __syncthreads();
  _Float16* h1b = &s_feat[w][0];
#pragma unroll
  for (int nt = 0; nt < 8; ++nt)
#pragma unroll
    for (int v = 0; v < 8; ++v)
      h1b[(v + 8 * hi) * HD + col0 + 16 * nt] = (_Float16)leakyf(acc[nt][v]);
  __syncthreads();

  // ---- GEMM2: 8 N-tiles x 4 K-chunks ----
#pragma unroll
  for (int nt = 0; nt < 8; ++nt) {
    float bv = b2g[nt * 16 + col0];
#pragma unroll
    for (int j = 0; j < 8; ++j) acc[nt][j] = bv;
  }
#pragma unroll
  for (int kk = 0; kk < 4; ++kk) {
    v16h a;
#pragma unroll
    for (int e = 0; e < 16; ++e) {
      int vv = e >> 1, h = e & 1;
      int kl = (2 * vv + h) + 16 * hi;
      a[e] = h1b[Mrow * HD + kk * 32 + kl];
    }
#pragma unroll
    for (int nt = 0; nt < 8; ++nt) {
      v16h bw = *((const v16h*)w2s + (nt * 4 + kk) * 32 + lane);
      acc[nt] = __builtin_amdgcn_wmma_f32_16x16x32_f16(
          false, a, false, bw, (short)0, acc[nt], false, false);
    }
  }
#pragma unroll
  for (int nt = 0; nt < 8; ++nt)
#pragma unroll
    for (int v = 0; v < 8; ++v) acc[nt][v] = leakyf(acc[nt][v]);

  // ---- WeightNet1 hidden (3->8->8), one neighbor per lane 0..15 ----
  if (lane < 16) {
    int id = s_idx[w][lane];
    float d0 = X2[id * 4 + 0] - X1[n * 4 + 0];
    float d1 = X2[id * 4 + 1] - X1[n * 4 + 1];
    float d2 = X2[id * 4 + 2] - X1[n * 4 + 2];
    float h1w[8];
#pragma unroll
    for (int dd = 0; dd < 8; ++dd)
      h1w[dd] = reluf(d0 * wn_w1[0 * 8 + dd] + d1 * wn_w1[1 * 8 + dd] + d2 * wn_w1[2 * 8 + dd] + wn_b1[dd]);
#pragma unroll
    for (int dd = 0; dd < 8; ++dd) {
      float s = wn_b2[dd];
#pragma unroll
      for (int e = 0; e < 8; ++e) s += h1w[e] * wn_w2[e * 8 + dd];
      s_wnh[w][lane][dd] = reluf(s);
    }
  }
  __syncthreads();

  // ---- point_to_patch: sum over 16 rows of relu(wn3)*H2 ----
#pragma unroll
  for (int nt = 0; nt < 8; ++nt) {
    int col = col0 + 16 * nt;
    float s = 0.f;
#pragma unroll
    for (int v = 0; v < 8; ++v) {
      int m = v + 8 * hi;
      float wv = wn_b3[col];
#pragma unroll
      for (int e = 0; e < 8; ++e) wv += s_wnh[w][m][e] * wn_w3[e * HD + col];
      s += reluf(wv) * acc[nt][v];
    }
    s += __shfl_xor(s, 16, 32);
    if (!hi) ptp[(size_t)g * HD + col] = s;
  }
}

// ---------------------------------------------------------------------------
// 6) Second grouping (rigid-velocity space): WeightNet2 + gather-reduce,
//    writes transposed output [B,128,N]. One block (128 threads) per point.
// ---------------------------------------------------------------------------
__global__ void ptp2_kernel(const float* __restrict__ x1t, const int* __restrict__ idx2,
                            const float* __restrict__ ptp,
                            const float* __restrict__ wn_w1, const float* __restrict__ wn_b1,
                            const float* __restrict__ wn_w2, const float* __restrict__ wn_b2,
                            const float* __restrict__ wn_w3, const float* __restrict__ wn_b3,
                            float* __restrict__ out) {
  __shared__ float s_h2[16][8];
  __shared__ int   s_id[16];
  int t = threadIdx.x;
  int g = blockIdx.x;
  int b = g / NN, n = g % NN;
  const float* X1 = x1t + (size_t)b * NN * 4;
  if (t < 16) {
    int id = idx2[(size_t)g * KNS + t];
    s_id[t] = id;
    float d0 = X1[id * 4 + 0] - X1[n * 4 + 0];
    float d1 = X1[id * 4 + 1] - X1[n * 4 + 1];
    float d2 = X1[id * 4 + 2] - X1[n * 4 + 2];
    float h1w[8];
#pragma unroll
    for (int dd = 0; dd < 8; ++dd)
      h1w[dd] = reluf(d0 * wn_w1[0 * 8 + dd] + d1 * wn_w1[1 * 8 + dd] + d2 * wn_w1[2 * 8 + dd] + wn_b1[dd]);
#pragma unroll
    for (int dd = 0; dd < 8; ++dd) {
      float s = wn_b2[dd];
#pragma unroll
      for (int e = 0; e < 8; ++e) s += h1w[e] * wn_w2[e * 8 + dd];
      s_h2[t][dd] = reluf(s);
    }
  }
  __syncthreads();
  const float* ptpb = ptp + (size_t)b * NN * HD;
  float acc = 0.f;
#pragma unroll
  for (int k = 0; k < KNS; ++k) {
    float wv = wn_b3[t];
#pragma unroll
    for (int e = 0; e < 8; ++e) wv += s_h2[k][e] * wn_w3[e * HD + t];
    acc += reluf(wv) * ptpb[(size_t)s_id[k] * HD + t];
  }
  out[(size_t)b * HD * NN + (size_t)t * NN + n] = acc;
}

// ---------------------------------------------------------------------------
extern "C" void kernel_launch(void* const* d_in, const int* in_sizes, int n_in,
                              void* d_out, int out_size, void* d_ws, size_t ws_size,
                              hipStream_t stream) {
  const float* xyz1    = (const float*)d_in[0];
  const float* xyz2    = (const float*)d_in[1];
  const float* points1 = (const float*)d_in[2];
  const float* points2 = (const float*)d_in[3];
  const float* vel1    = (const float*)d_in[4];
  const float* w_xyz   = (const float*)d_in[8];
  const float* w_points= (const float*)d_in[10];
  const float* mlp_w1  = (const float*)d_in[11];
  const float* mlp_b1  = (const float*)d_in[12];
  const float* mlp_w2  = (const float*)d_in[13];
  const float* mlp_b2  = (const float*)d_in[14];
  const float* wn1_w1  = (const float*)d_in[15];
  const float* wn1_b1  = (const float*)d_in[16];
  const float* wn1_w2  = (const float*)d_in[17];
  const float* wn1_b2  = (const float*)d_in[18];
  const float* wn1_w3  = (const float*)d_in[19];
  const float* wn1_b3  = (const float*)d_in[20];
  const float* wn2_w1  = (const float*)d_in[21];
  const float* wn2_b1  = (const float*)d_in[22];
  const float* wn2_w2  = (const float*)d_in[23];
  const float* wn2_b2  = (const float*)d_in[24];
  const float* wn2_w3  = (const float*)d_in[25];
  const float* wn2_b3  = (const float*)d_in[26];

  float* out = (float*)d_out;
  char* ws = (char*)d_ws;
  size_t off = 0;
  auto alloc = [&](size_t bytes) { size_t o = off; off = (off + bytes + 255) & ~(size_t)255; return o; };

  float*    ft1  = (float*)(ws + alloc((size_t)BB * NN * FPAD * 4));
  float*    ft2  = (float*)(ws + alloc((size_t)BB * NN * FPAD * 4));
  float*    x1t  = (float*)(ws + alloc((size_t)BB * NN * 4 * 4));
  float*    x2t  = (float*)(ws + alloc((size_t)BB * NN * 4 * 4));
  float*    vw   = (float*)(ws + alloc((size_t)BB * NN * 4 * 4));
  int*      idxc = (int*)  (ws + alloc((size_t)BB * NN * KCL * 4));
  int*      knni = (int*)  (ws + alloc((size_t)BB * NN * KNS * 4));
  int*      idx2 = (int*)  (ws + alloc((size_t)BB * NN * KNS * 4));
  float*    ptp  = (float*)(ws + alloc((size_t)BB * NN * HD * 4));
  _Float16* w1s  = (_Float16*)(ws + alloc((size_t)W1TOT * 2));
  _Float16* w2s  = (_Float16*)(ws + alloc((size_t)W2TOT * 2));

  float* out_tail = out + (size_t)BB * HD * NN;   // v_world output region

  prep_kernel<<<(BB * NN + 255) / 256, 256, 0, stream>>>(
      xyz1, xyz2, points1, points2, w_xyz, w_points, ft1, ft2, x1t, x2t);

  swizzle_w_kernel<<<(W1TOT + W2TOT + 255) / 256, 256, 0, stream>>>(
      mlp_w1, mlp_w2, w1s, w2s);

  // cluster knn (xyz self, k=8)
  knn_kernel<<<BB * NN, 256, 0, stream>>>(x1t, x1t, 4, 4, KCL, idxc);

  vworld_kernel<<<(BB * NN + 255) / 256, 256, 0, stream>>>(x1t, vel1, idxc, vw, out_tail);

  // feature-space knn (67-dim padded to 68, k=16)
  knn_kernel<<<BB * NN, 256, 0, stream>>>(ft1, ft2, FPAD, FPAD, KNS, knni);

  // WMMA cost-volume MLP + WeightNet1 reduction
  mlp_ptp_kernel<<<BB * NN / 8, 256, 0, stream>>>(
      points1, points2, x1t, x2t, knni, w1s, w2s, mlp_b1, mlp_b2,
      wn1_w1, wn1_b1, wn1_w2, wn1_b2, wn1_w3, wn1_b3, ptp);

  // rigid-velocity-space knn (k=16)
  knn_kernel<<<BB * NN, 256, 0, stream>>>(vw, vw, 4, 4, KNS, idx2);

  // second grouping + transpose-out
  ptp2_kernel<<<BB * NN, 128, 0, stream>>>(
      x1t, idx2, ptp, wn2_w1, wn2_b1, wn2_w2, wn2_b2, wn2_w3, wn2_b3, out);
}
